// GraphHINGE_CNN_63178968924633
// MI455X (gfx1250) — matrix operand order repaired and checked by reading, not compiled, not run on hardware
//
#include <hip/hip_runtime.h>

// GraphHINGE forward for MI455X (gfx1250): f16 WMMA GEMM core + VALU reductions.
// v3: B matrices pre-swizzled to WMMA fragment order (1 contiguous 32B read
// per lane per fragment), C tiles staged through LDS for coalesced b128
// stores, const-division gather addressing.
#define BB     256
#define LLEN   64
#define EE     128
#define NHEAD  3
#define BLROWS (BB * LLEN)   // 16384
#define TEMP1  0.2f
#define TEMP2  0.2f

typedef _Float16 half_t;
typedef __attribute__((ext_vector_type(16))) _Float16 v16h;
typedef __attribute__((ext_vector_type(8)))  _Float16 v8h;
typedef __attribute__((ext_vector_type(8)))  float    v8f;

struct Tables { const float* t[5]; };

// ---- WMMA fragment helpers (layouts per CDNA5 ISA 7.12.2, wave32) ----------

// A: 16x32 f16 tile staged row-major in LDS.
// lane<16: M=lane, K in {0..7,16..23}; lane>=16: M=lane-16, K in {8..15,24..31}
__device__ __forceinline__ v16h frag_a_lds(const half_t* lds_a, int lane) {
  const int row = lane & 15;
  const int kb  = (lane < 16) ? 0 : 8;
  v16h a;
#pragma unroll
  for (int i = 0; i < 16; ++i) {
    const int k = kb + ((i < 8) ? i : (i + 8));
    a[i] = lds_a[row * 32 + k];
  }
  return a;
}

// A fragment straight from global f16 row-major (two contiguous 16B runs/lane).
__device__ __forceinline__ v16h frag_a_gl(const half_t* __restrict__ A,
                                          long row_stride, int mbase, int kofs,
                                          int lane) {
  const int row = mbase + (lane & 15);
  const int kb  = (lane < 16) ? 0 : 8;
  const half_t* p = A + (long)row * row_stride + kofs + kb;
  v8h lo = *(const v8h*)p;
  v8h hi = *(const v8h*)(p + 16);
  return __builtin_shufflevector(lo, hi, 0, 1, 2, 3, 4, 5, 6, 7,
                                 8, 9, 10, 11, 12, 13, 14, 15);
}

// B fragment from pre-swizzled buffer: 32 contiguous bytes per lane.
// Layout: sw[((kc*8+tn)*32+lane)*16 + i]
__device__ __forceinline__ v16h frag_b_sw(const half_t* __restrict__ Bsw,
                                          int kc, int tn, int lane) {
  return *(const v16h*)(Bsw + (((long)(kc * 8 + tn) * 32 + lane) << 4));
}

__device__ __forceinline__ v8f wmma_f16(v16h a, v16h b, v8f c) {
  return __builtin_amdgcn_wmma_f32_16x16x32_f16(false, a, false, b, (short)0, c,
                                                false, false);
}

// ---- weight conversion + swizzle -------------------------------------------

// conv_w (5,E,E,2) f32 -> per-path swizzled (2E x E) f16 B-fragments (K=256).
__global__ void k_conv_cat_sw(const float* __restrict__ cw,
                              half_t* __restrict__ dst) {
  long gi = (long)blockIdx.x * blockDim.x + threadIdx.x;
  if (gi >= 5L * 32768) return;
  const int p = (int)(gi >> 15);
  const int r = (int)(gi & 32767);
  const int i = r & 15, lane = (r >> 4) & 31, tn = (r >> 9) & 7, kc = r >> 12;
  const int k = kc * 32 + ((lane < 16) ? 0 : 8) + ((i < 8) ? i : (i + 8));
  const int o = tn * 16 + (lane & 15);
  const int e = k & 127, slot = k >> 7;
  dst[gi] = (half_t)cw[(long)p * 32768 + o * 256 + e * 2 + slot];
}

// nmat K-major (128x128) f32 matrices -> swizzled f16 B-fragments (K=128).
__global__ void k_swz_b128(const float* __restrict__ src,
                           half_t* __restrict__ dst, int nmat) {
  long gi = (long)blockIdx.x * blockDim.x + threadIdx.x;
  if (gi >= (long)nmat * 16384) return;
  const int m = (int)(gi >> 14);
  const int r = (int)(gi & 16383);
  const int i = r & 15, lane = (r >> 4) & 31, tn = (r >> 9) & 7, kc = r >> 12;
  const int k = kc * 32 + ((lane < 16) ? 0 : 8) + ((i < 8) ? i : (i + 8));
  const int col = tn * 16 + (lane & 15);
  dst[gi] = (half_t)src[(long)m * 16384 + k * 128 + col];
}

// ---- fused gather + meta_embed GEMM: (BL*np x 2E) @ (2E x E) + bias --------
// One wave -> full 16(M) x 128(N) tile: 8 WMMA per K-step (8 steps).

__global__ void k_meta_gemm(Tables tabs, const int* __restrict__ idx, int n_nodes,
                            int t0, int t1, int t2, int t3,
                            const half_t* __restrict__ Wsw /*swizzled 256x128*/,
                            const float* __restrict__ bias,
                            half_t* __restrict__ Out) {
  __shared__ half_t lds_a[16 * 32];
  __shared__ half_t lds_o[16 * 128];
  const int lane = threadIdx.x;
  const int tm = blockIdx.x;
  const int np = n_nodes - 1;
  int ntab[4] = {t0, t1, t2, t3};
  __builtin_prefetch(Wsw, 0, 1);
  v8f acc[8];
  const v8f vzero = {};
#pragma unroll
  for (int tn = 0; tn < 8; ++tn) acc[tn] = vzero;
#pragma unroll 1
  for (int kc = 0; kc < 8; ++kc) {           // K = 256 in steps of 32
    const int node_off = (kc < 4) ? 0 : 1;
    const int eofs = (kc & 3) * 32;
    for (int r = 0; r < 16; ++r) {           // cooperative coalesced gather -> LDS
      const int m  = tm * 16 + r;
      const int bl = (np == 3) ? (m / 3) : m;  // const-division
      const int j  = m - bl * np;
      const int node = j + node_off;
      const int id = idx[(long)bl * n_nodes + node];
      const float* src = tabs.t[ntab[node]] + (long)id * EE + eofs;
      lds_a[r * 32 + lane] = (half_t)src[lane];
    }
    __syncthreads();
    v16h a = frag_a_lds(lds_a, lane);
#pragma unroll
    for (int tn = 0; tn < 8; ++tn)
      acc[tn] = wmma_f16(a, frag_b_sw(Wsw, kc, tn, lane), acc[tn]);
    __syncthreads();
  }
  // bias + pack tile into LDS, then coalesced b128 writeback
  const int col = lane & 15;
  const int rbase = (lane < 16) ? 0 : 8;
#pragma unroll
  for (int tn = 0; tn < 8; ++tn) {
    const float bv = bias[tn * 16 + col];
#pragma unroll
    for (int r = 0; r < 8; ++r)
      lds_o[(rbase + r) * EE + tn * 16 + col] = (half_t)(acc[tn][r] + bv);
  }
  __syncthreads();
  half_t* outp = Out + (long)tm * 16 * EE;
#pragma unroll
  for (int it = 0; it < 8; ++it)
    *(v8h*)(outp + it * 256 + lane * 8) = *(const v8h*)(lds_o + it * 256 + lane * 8);
}

// ---- fused Ws/Wc projection: (M x 128) @ two (128 x 128), per-head grid.z --
// One wave -> 16 x 128 for BOTH outputs: 16 WMMA per K-step.

__global__ void k_proj_gemm(const half_t* __restrict__ A,
                            const half_t* __restrict__ Bs,
                            const half_t* __restrict__ Bc,
                            half_t* __restrict__ hj, half_t* __restrict__ hc,
                            long out_head_stride) {
  __shared__ half_t lds_o[16 * 128];
  const int lane = threadIdx.x;
  const int tm = blockIdx.x, h = blockIdx.z;
  const half_t* Bsh = Bs + (long)h * EE * EE;
  const half_t* Bch = Bc + (long)h * EE * EE;
  __builtin_prefetch(Bsh, 0, 1);
  __builtin_prefetch(Bch, 0, 1);
  half_t* hjh = hj + (long)h * out_head_stride;
  half_t* hch = hc + (long)h * out_head_stride;
  v8f accs[8], accc[8];
  const v8f vzero = {};
#pragma unroll
  for (int tn = 0; tn < 8; ++tn) { accs[tn] = vzero; accc[tn] = vzero; }
#pragma unroll 1
  for (int kc = 0; kc < 4; ++kc) {           // K = 128
    v16h a = frag_a_gl(A, EE, tm * 16, kc * 32, lane);
#pragma unroll
    for (int tn = 0; tn < 8; ++tn) {
      accs[tn] = wmma_f16(a, frag_b_sw(Bsh, kc, tn, lane), accs[tn]);
      accc[tn] = wmma_f16(a, frag_b_sw(Bch, kc, tn, lane), accc[tn]);
    }
  }
  const int col = lane & 15;
  const int rbase = (lane < 16) ? 0 : 8;
#pragma unroll
  for (int tn = 0; tn < 8; ++tn)
#pragma unroll
    for (int r = 0; r < 8; ++r)
      lds_o[(rbase + r) * EE + tn * 16 + col] = (half_t)accs[tn][r];
  __syncthreads();
  half_t* oj = hjh + (long)tm * 16 * EE;
#pragma unroll
  for (int it = 0; it < 8; ++it)
    *(v8h*)(oj + it * 256 + lane * 8) = *(const v8h*)(lds_o + it * 256 + lane * 8);
  __syncthreads();
#pragma unroll
  for (int tn = 0; tn < 8; ++tn)
#pragma unroll
    for (int r = 0; r < 8; ++r)
      lds_o[(rbase + r) * EE + tn * 16 + col] = (half_t)accc[tn][r];
  __syncthreads();
  half_t* oc = hch + (long)tm * 16 * EE;
#pragma unroll
  for (int it = 0; it < 8; ++it)
    *(v8h*)(oc + it * 256 + lane * 8) = *(const v8h*)(lds_o + it * 256 + lane * 8);
}

// ---- generic single-B GEMM (used for hi; strided A rows) -------------------

__global__ void k_gemm128(const half_t* __restrict__ A, long a_row_stride,
                          const half_t* __restrict__ Bw, long b_head_stride,
                          half_t* __restrict__ Out, long out_head_stride) {
  __shared__ half_t lds_o[16 * 128];
  const int lane = threadIdx.x;
  const int tm = blockIdx.x, h = blockIdx.z;
  const half_t* Bh = Bw + (long)h * b_head_stride;
  __builtin_prefetch(Bh, 0, 1);
  half_t* Oh = Out + (long)h * out_head_stride;
  v8f acc[8];
  const v8f vzero = {};
#pragma unroll
  for (int tn = 0; tn < 8; ++tn) acc[tn] = vzero;
#pragma unroll 1
  for (int kc = 0; kc < 4; ++kc) {
    v16h a = frag_a_gl(A, a_row_stride, tm * 16, kc * 32, lane);
#pragma unroll
    for (int tn = 0; tn < 8; ++tn)
      acc[tn] = wmma_f16(a, frag_b_sw(Bh, kc, tn, lane), acc[tn]);
  }
  const int col = lane & 15;
  const int rbase = (lane < 16) ? 0 : 8;
#pragma unroll
  for (int tn = 0; tn < 8; ++tn)
#pragma unroll
    for (int r = 0; r < 8; ++r)
      lds_o[(rbase + r) * EE + tn * 16 + col] = (half_t)acc[tn][r];
  __syncthreads();
  half_t* outp = Oh + (long)tm * 16 * EE;
#pragma unroll
  for (int it = 0; it < 8; ++it)
    *(v8h*)(outp + it * 256 + lane * 8) = *(const v8h*)(lds_o + it * 256 + lane * 8);
}

// ---- node attention: logits dot, softmax over np, head-mean weighted sum ---

__global__ void k_node_att(const half_t* __restrict__ hi /*NH,BL,E*/,
                           const half_t* __restrict__ hj /*NH,BL*np,E*/,
                           const half_t* __restrict__ hc, int np,
                           float* __restrict__ Zout /*BL,E*/) {
  const int bl = blockIdx.x, lane = threadIdx.x;
  float acc[4] = {0.f, 0.f, 0.f, 0.f};
  for (int h = 0; h < NHEAD; ++h) {
    const half_t* hiv = hi + ((long)h * BLROWS + bl) * EE;
    float logit[3];
    for (int n = 0; n < np; ++n) {
      const half_t* hjv = hj + ((long)h * BLROWS * np + (long)bl * np + n) * EE;
      float part = 0.f;
      for (int f = lane; f < EE; f += 32) part += (float)hjv[f] * (float)hiv[f];
      for (int o = 16; o > 0; o >>= 1) part += __shfl_xor(part, o, 32);
      logit[n] = part * (1.0f / TEMP1);
    }
    float mx = logit[0];
    for (int n = 1; n < np; ++n) mx = fmaxf(mx, logit[n]);
    float al[3], den = 0.f;
    for (int n = 0; n < np; ++n) { al[n] = __expf(logit[n] - mx); den += al[n]; }
    const float inv = 1.0f / den;
    for (int n = 0; n < np; ++n) {
      const half_t* hcv = hc + ((long)h * BLROWS * np + (long)bl * np + n) * EE;
      const float w = al[n] * inv * (1.0f / (float)NHEAD);
#pragma unroll
      for (int q = 0; q < 4; ++q) acc[q] += w * (float)hcv[lane * 4 + q];
    }
  }
#pragma unroll
  for (int q = 0; q < 4; ++q) Zout[(long)bl * EE + lane * 4 + q] = acc[q];
}

// ---- path attention ---------------------------------------------------------

__global__ void k_path_scores(const float* __restrict__ Zside /*5,BL,E*/,
                              const float* __restrict__ pw1,
                              const float* __restrict__ pb1,
                              const float* __restrict__ pw2,
                              float* __restrict__ scores /*B,320*/) {
  __shared__ float zrow[EE];
  const int row = blockIdx.x, lane = threadIdx.x;
  const int b = row / 320, q = row - b * 320;
  const int path = q >> 6, l = q & 63;
  const float* z = Zside + ((long)path * BLROWS + b * LLEN + l) * EE;
  for (int e = lane; e < EE; e += 32) zrow[e] = z[e];
  __syncthreads();
  float s = 0.f;
  for (int f0 = 0; f0 < EE; f0 += 32) {
    const int f = f0 + lane;
    float a = pb1[f];
    for (int e = 0; e < EE; ++e) a += zrow[e] * pw1[e * EE + f];
    s += tanhf(a) * pw2[f];
  }
  for (int o = 16; o > 0; o >>= 1) s += __shfl_xor(s, o, 32);
  if (lane == 0) scores[row] = s * (1.0f / TEMP2);
}

__global__ void k_path_sum(const float* __restrict__ Zside,
                           const float* __restrict__ scores,
                           float* __restrict__ out /*B,E*/) {
  __shared__ float red[EE];
  const int b = blockIdx.x, t = threadIdx.x;
  float mx = -1e30f;
  for (int q = t; q < 320; q += EE) mx = fmaxf(mx, scores[b * 320 + q]);
  red[t] = mx; __syncthreads();
  for (int s = 64; s > 0; s >>= 1) { if (t < s) red[t] = fmaxf(red[t], red[t + s]); __syncthreads(); }
  mx = red[0]; __syncthreads();
  float sum = 0.f;
  for (int q = t; q < 320; q += EE) sum += __expf(scores[b * 320 + q] - mx);
  red[t] = sum; __syncthreads();
  for (int s = 64; s > 0; s >>= 1) { if (t < s) red[t] += red[t + s]; __syncthreads(); }
  const float inv = 1.0f / red[0];
  float acc = 0.f;
  for (int q = 0; q < 320; ++q) {
    const float beta = __expf(scores[b * 320 + q] - mx) * inv;
    const int path = q >> 6, l = q & 63;
    acc += beta * Zside[((long)path * BLROWS + b * LLEN + l) * EE + t];
  }
  out[(long)b * EE + t] = acc;
}

// ---- final MLP --------------------------------------------------------------

__global__ void k_final(const float* __restrict__ ue, const float* __restrict__ ie,
                        const int* __restrict__ UIUI, const int* __restrict__ IUIU,
                        const float* __restrict__ up, const float* __restrict__ ip,
                        const float* __restrict__ fw1, const float* __restrict__ fb1,
                        const float* __restrict__ fw2, const float* __restrict__ fb2,
                        float* __restrict__ out) {
  __shared__ float x[4 * EE];
  __shared__ float red[EE];
  const int b = blockIdx.x, t = threadIdx.x;
  const int sid = UIUI[(long)b * LLEN * 4];
  const int tid = IUIU[(long)b * LLEN * 4];
  x[t]          = ue[(long)sid * EE + t];
  x[EE + t]     = ie[(long)tid * EE + t];
  x[2 * EE + t] = up[(long)b * EE + t];
  x[3 * EE + t] = ip[(long)b * EE + t];
  __syncthreads();
  float a = fb1[t];
  for (int k = 0; k < 4 * EE; ++k) a += x[k] * fw1[(long)k * EE + t];
  red[t] = fmaxf(a, 0.f) * fw2[t];
  __syncthreads();
  for (int s = 64; s > 0; s >>= 1) { if (t < s) red[t] += red[t + s]; __syncthreads(); }
  if (t == 0) out[b] = 1.0f / (1.0f + __expf(-(red[0] + fb2[0])));
}

// ---- host orchestration -----------------------------------------------------

extern "C" void kernel_launch(void* const* d_in, const int* in_sizes, int n_in,
                              void* d_out, int out_size, void* d_ws, size_t ws_size,
                              hipStream_t stream) {
  (void)in_sizes; (void)n_in; (void)out_size; (void)ws_size;
  const int* idxs[10];
  for (int i = 0; i < 10; ++i) idxs[i] = (const int*)d_in[i];
  Tables tabs;
  for (int i = 0; i < 5; ++i) tabs.t[i] = (const float*)d_in[10 + i];
  const float* conv_w[2] = {(const float*)d_in[15], (const float*)d_in[17]};
  const float* conv_b[2] = {(const float*)d_in[16], (const float*)d_in[18]};
  const float* Wt[2] = {(const float*)d_in[19], (const float*)d_in[22]};
  const float* Ws[2] = {(const float*)d_in[20], (const float*)d_in[23]};
  const float* Wc[2] = {(const float*)d_in[21], (const float*)d_in[24]};
  const float* pw1[2] = {(const float*)d_in[25], (const float*)d_in[28]};
  const float* pb1[2] = {(const float*)d_in[26], (const float*)d_in[29]};
  const float* pw2[2] = {(const float*)d_in[27], (const float*)d_in[30]};
  const float* fw1 = (const float*)d_in[31];
  const float* fb1 = (const float*)d_in[32];
  const float* fw2 = (const float*)d_in[33];
  const float* fb2 = (const float*)d_in[34];

  char* ws = (char*)d_ws;
  size_t off = 0;
  auto take = [&](size_t bytes) -> char* {
    char* p = ws + off;
    off += (bytes + 255) & ~(size_t)255;
    return p;
  };
  const size_t ATTW = 5L * NHEAD * EE * EE;          // per-side att weight halfs
  half_t* convW16 = (half_t*)take(2L * 5 * 256 * EE * sizeof(half_t));
  half_t* attWt16 = (half_t*)take(2L * ATTW * sizeof(half_t));
  half_t* attWs16 = (half_t*)take(2L * ATTW * sizeof(half_t));
  half_t* attWc16 = (half_t*)take(2L * ATTW * sizeof(half_t));
  half_t* metaBuf = (half_t*)take((size_t)BLROWS * 3 * EE * sizeof(half_t));
  half_t* hjBuf   = (half_t*)take((size_t)NHEAD * BLROWS * 3 * EE * sizeof(half_t));
  half_t* hcBuf   = (half_t*)take((size_t)NHEAD * BLROWS * 3 * EE * sizeof(half_t));
  half_t* hiBuf   = (half_t*)take((size_t)NHEAD * BLROWS * EE * sizeof(half_t));
  float*  Zbuf    = (float*)take(10L * BLROWS * EE * sizeof(float));
  float*  scBuf   = (float*)take(2L * BB * 320 * sizeof(float));
  float*  upip    = (float*)take(2L * BB * EE * sizeof(float));

  // 1) weight conversion + swizzle into WMMA B-fragment order
  for (int s = 0; s < 2; ++s) {
    k_conv_cat_sw<<<(5 * 32768 + 255) / 256, 256, 0, stream>>>(
        conv_w[s], convW16 + (size_t)s * 5 * 32768);
    k_swz_b128<<<((int)ATTW + 255) / 256, 256, 0, stream>>>(Wt[s], attWt16 + s * ATTW, 15);
    k_swz_b128<<<((int)ATTW + 255) / 256, 256, 0, stream>>>(Ws[s], attWs16 + s * ATTW, 15);
    k_swz_b128<<<((int)ATTW + 255) / 256, 256, 0, stream>>>(Wc[s], attWc16 + s * ATTW, 15);
  }

  // path metadata: node->table ids and index-input slot, per side
  static const int utab[5][4] = {{0,1,0,0},{0,1,0,1},{0,1,2,1},{0,1,3,1},{0,1,4,1}};
  static const int itab[5][4] = {{1,0,0,0},{1,0,1,0},{1,2,1,0},{1,3,1,0},{1,4,1,0}};
  static const int uix[5] = {0, 2, 4, 6, 8};
  static const int iix[5] = {1, 3, 5, 7, 9};

  // 2) per-path: meta GEMM -> hj/hc fused GEMM -> hi GEMM -> node attention
  for (int s = 0; s < 2; ++s) {
    for (int p = 0; p < 5; ++p) {
      const int np = (p == 0) ? 1 : 3;
      const int nn = np + 1;
      const int M = BLROWS * np;
      const int* tb = s ? itab[p] : utab[p];
      const int* ix = idxs[s ? iix[p] : uix[p]];
      const half_t* Wsw = convW16 + ((size_t)s * 5 + p) * 32768;
      const float* bias = conv_b[s] + (size_t)p * EE;

      k_meta_gemm<<<M / 16, 32, 0, stream>>>(
          tabs, ix, nn, tb[0], tb[1], tb[2], tb[3], Wsw, bias, metaBuf);

      const size_t wofs = ((size_t)s * 5 + p) * NHEAD * EE * EE;
      k_proj_gemm<<<dim3(M / 16, 1, NHEAD), 32, 0, stream>>>(
          metaBuf, attWs16 + wofs, attWc16 + wofs, hjBuf, hcBuf, (long)M * EE);
      k_gemm128<<<dim3(BLROWS / 16, 1, NHEAD), 32, 0, stream>>>(
          metaBuf, (long)np * EE, attWt16 + wofs, (long)EE * EE, hiBuf,
          (long)BLROWS * EE);

      k_node_att<<<BLROWS, 32, 0, stream>>>(
          hiBuf, hjBuf, hcBuf, np, Zbuf + ((size_t)s * 5 + p) * BLROWS * EE);
    }
  }

  // 3) path attention per side
  for (int s = 0; s < 2; ++s) {
    const float* Zside = Zbuf + (size_t)s * 5 * BLROWS * EE;
    k_path_scores<<<BB * 320, 32, 0, stream>>>(Zside, pw1[s], pb1[s], pw2[s],
                                               scBuf + (size_t)s * BB * 320);
    k_path_sum<<<BB, EE, 0, stream>>>(Zside, scBuf + (size_t)s * BB * 320,
                                      upip + (size_t)s * BB * EE);
  }

  // 4) final MLP
  k_final<<<BB, EE, 0, stream>>>((const float*)d_in[10], (const float*)d_in[11],
                                 idxs[2], idxs[3], upip, upip + (size_t)BB * EE,
                                 fw1, fb1, fw2, fb2, (float*)d_out);
}
